// PointNetFPModule_60610578481761
// MI455X (gfx1250) — matrix operand order repaired and checked by reading, not compiled, not run on hardware
//
#include <hip/hip_runtime.h>

// Problem constants (match reference)
#define B_    8
#define N_    8192
#define M_    1024
#define C1_   256
#define C2_   512
#define CIN_  768
#define CMID_ 256
#define COUT_ 256
#define L_    (B_ * N_)          // 65536 total points

typedef __attribute__((ext_vector_type(16))) _Float16 v16h;
typedef __attribute__((ext_vector_type(8)))  float    v8f;
typedef __attribute__((ext_vector_type(4)))  unsigned int v4u;
typedef __attribute__((ext_vector_type(8)))  int v8i;
typedef __attribute__((ext_vector_type(4)))  int v4i;

union FragU { v16h v; uint4 q[2]; };
union HalfU { uint4 q[2]; _Float16 h[16]; };

// Tensor Data Mover availability (device pass on gfx1250 toolchains has it;
// host pass / older toolchains fall back to the proven manual staging path).
#if defined(__has_builtin)
#if __has_builtin(__builtin_amdgcn_tensor_load_to_lds)
#define USE_TDM 1
#else
#define USE_TDM 0
#endif
#else
#define USE_TDM 0
#endif

// ---------------------------------------------------------------------------
// Zero BN accumulators (sum1, sq1, sum2, sq2 = 4*256 floats)
// ---------------------------------------------------------------------------
__global__ void zero_stats_kernel(float* __restrict__ stats) {
  int i = blockIdx.x * blockDim.x + threadIdx.x;
  if (i < 1024) stats[i] = 0.0f;
}

// ---------------------------------------------------------------------------
// 3-NN: for each unknown point, find 3 nearest known points + IDW weights.
// One block = 256 query points of one batch; known xyz staged in LDS (12KB).
// ---------------------------------------------------------------------------
__global__ __launch_bounds__(256) void three_nn_kernel(
    const float* __restrict__ unknown,   // (B, N, 3)
    const float* __restrict__ known,     // (B, M, 3)
    int* __restrict__ nn_idx,            // (L, 3)
    float* __restrict__ nn_w) {          // (L, 3)
  __shared__ float kx[M_], ky[M_], kz[M_];
  const int b = blockIdx.x >> 5;                       // 32 blocks per batch
  const int n = ((blockIdx.x & 31) << 8) + threadIdx.x;
  const float* kb = known + (size_t)b * M_ * 3;
  for (int j = threadIdx.x; j < M_; j += 256) {
    kx[j] = kb[j * 3 + 0];
    ky[j] = kb[j * 3 + 1];
    kz[j] = kb[j * 3 + 2];
  }
  __syncthreads();
  const float* up = unknown + ((size_t)b * N_ + n) * 3;
  const float ux = up[0], uy = up[1], uz = up[2];
  float d0 = 1e30f, d1 = 1e30f, d2 = 1e30f;
  int   i0 = 0, i1 = 0, i2 = 0;
  for (int j = 0; j < M_; ++j) {
    const float dx = ux - kx[j], dy = uy - ky[j], dz = uz - kz[j];
    const float d = fmaf(dx, dx, fmaf(dy, dy, dz * dz));
    if (d < d2) {
      if (d < d0)      { d2 = d1; i2 = i1; d1 = d0; i1 = i0; d0 = d; i0 = j; }
      else if (d < d1) { d2 = d1; i2 = i1; d1 = d;  i1 = j; }
      else             { d2 = d;  i2 = j; }
    }
  }
  const float w0 = 1.0f / (d0 + 1e-8f);
  const float w1 = 1.0f / (d1 + 1e-8f);
  const float w2 = 1.0f / (d2 + 1e-8f);
  const float inv = 1.0f / (w0 + w1 + w2);
  const int l = b * N_ + n;
  nn_idx[l * 3 + 0] = i0; nn_idx[l * 3 + 1] = i1; nn_idx[l * 3 + 2] = i2;
  nn_w[l * 3 + 0] = w0 * inv; nn_w[l * 3 + 1] = w1 * inv; nn_w[l * 3 + 2] = w2 * inv;
}

// ---------------------------------------------------------------------------
// Transposed copy: unknow_feats (B, C1, N) fp32 -> Xcat[L][0..C1) f16
// ---------------------------------------------------------------------------
__global__ __launch_bounds__(256) void copy_transpose_kernel(
    const float* __restrict__ uf,        // (B, C1, N)
    _Float16* __restrict__ Xcat) {       // [L][CIN]
  __shared__ float tile[32][33];
  const int lb = blockIdx.x * 32;
  const int cb = blockIdx.y * 32;
  const int tx = threadIdx.x, ty = threadIdx.y;
  const int b = lb >> 13;                // N_ = 8192
  const int nb = lb & (N_ - 1);
#pragma unroll
  for (int i = 0; i < 32; i += 8)
    tile[ty + i][tx] = uf[((size_t)(b * C1_ + cb + ty + i)) * N_ + nb + tx];
  __syncthreads();
#pragma unroll
  for (int i = 0; i < 32; i += 8)
    Xcat[(size_t)(lb + ty + i) * CIN_ + cb + tx] = (_Float16)tile[tx][ty + i];
}

// ---------------------------------------------------------------------------
// Interpolation: Xcat[l][C1 + c] = sum_k w[l][k] * known_feats[b][c][idx[l][k]]
// Gathers are random within 2MB/batch -> L2-resident (192MB L2).
// ---------------------------------------------------------------------------
__global__ __launch_bounds__(256) void interp_concat_kernel(
    const float* __restrict__ kf,        // (B, C2, M)
    const int* __restrict__ nn_idx,
    const float* __restrict__ nn_w,
    _Float16* __restrict__ Xcat) {       // [L][CIN]
  const int l = blockIdx.x;
  const int b = l >> 13;
  const int i0 = nn_idx[l * 3 + 0], i1 = nn_idx[l * 3 + 1], i2 = nn_idx[l * 3 + 2];
  const float w0 = nn_w[l * 3 + 0], w1 = nn_w[l * 3 + 1], w2 = nn_w[l * 3 + 2];
  const float* kfb = kf + (size_t)b * C2_ * M_;
  _Float16* xr = Xcat + (size_t)l * CIN_ + C1_;
#pragma unroll
  for (int c = threadIdx.x; c < C2_; c += 256) {
    const float* row = kfb + (size_t)c * M_;
    const float v = fmaf(w0, row[i0], fmaf(w1, row[i1], w2 * row[i2]));
    xr[c] = (_Float16)v;
  }
}

// ---------------------------------------------------------------------------
// WMMA GEMM:  Y[l][o] = sum_k A[l][k] * W[o][k]    (A f16, W f32->f16)
// Block tile 128L x 128O, 8 waves (4L x 2O), wave tile 32L x 64O = 8 WMMAs/step.
// A tile is DMA'd global->LDS by the Tensor Data Mover (one tensor_load_to_lds
// per workgroup per K-step); TDM LDS-padding (12 dwords every 16 dwords)
// produces a 56-half row pitch: 16B-aligned, conflict-free ds_load_b128.
// W tile is staged by VALU (needs f32->f16 conversion).
// Accumulates per-channel sum / sum^2 for training-mode BN stats.
// Dynamic LDS layout: As @ 0 (14336B), Ws @ 14336 (14336B), bsum/bsq @ 28672.
// ---------------------------------------------------------------------------
#define GEMM_LDS_BYTES (14336 + 14336 + 1024)

template <bool OUT_HALF>
__global__ __launch_bounds__(256) void gemm_wmma_kernel(
    const _Float16* __restrict__ A,      // [L][K]
    const float* __restrict__ W,         // [Otot][K] fp32
    const int K,
    void* __restrict__ Yout,             // [L][Otot] f16 or f32 (pre-BN)
    const int Otot,
    float* __restrict__ sumO,            // [Otot]
    float* __restrict__ sqO) {           // [Otot]
  constexpr int PITCH = 56;              // halfs; 112B row pitch
  extern __shared__ __attribute__((aligned(16))) char smem[];
  _Float16* As = (_Float16*)smem;                 // offset 0 (TDM target)
  _Float16* Ws = (_Float16*)(smem + 14336);
  float* bsum = (float*)(smem + 28672);
  float* bsq  = bsum + 128;

  const int tid  = threadIdx.x;
  const int lane = tid & 31;
  const int wave = tid >> 5;
  const int wl = wave & 3;               // wave row (L): 0..3, 32 L each
  const int wo = wave >> 2;              // wave col (O): 0..1, 64 O each
  const int h  = lane >> 4;              // half-wave
  const int ln = lane & 15;

  const int lb0 = blockIdx.x * 128;
  const int ob0 = blockIdx.y * 128;

  if (tid < 128) { bsum[tid] = 0.0f; bsq[tid] = 0.0f; }

  v8f acc[2][4];
#pragma unroll
  for (int lt = 0; lt < 2; ++lt)
#pragma unroll
    for (int ot = 0; ot < 4; ++ot)
#pragma unroll
      for (int r = 0; r < 8; ++r) acc[lt][ot][r] = 0.0f;

  const int rowS = tid >> 1;             // 0..127 (staging row)
  const int colh = (tid & 1) * 16;       // 0 or 16 (staging half-row)

  for (int k0 = 0; k0 < K; k0 += 32) {
#if USE_TDM
    // ---- A tile via Tensor Data Mover: 128 rows x 32 halfs, LDS-padded ----
    if (wave == 0) {
      const unsigned long long ga =
          (unsigned long long)(const char*)(A + (size_t)lb0 * K + k0);
      const unsigned ldsa = (unsigned)(size_t)As;   // low 32 bits = LDS offset
      // D# group0: count=1 | lds_addr | global_addr[56:0] | type=2
      v4u g0 = {1u, ldsa, (unsigned)ga,
                (unsigned)((ga >> 32) & 0x1ffffffull) | 0x80000000u};
      // D# group1: data_size=2B, pad_enable, pad_interval=16dw, pad_amount=12dw
      //            tensor 32 x 128, tile 32 x 128, dim0_stride = K
      const unsigned d0 = (1u << 16) | (1u << 20) | (3u << 22) | (11u << 25);
      v8i g1 = {(int)d0,
                (int)(32u << 16),                 // tensor_dim0 = 32
                (int)(128u << 16),                // tensor_dim1 = 128
                (int)(32u << 16),                 // tile_dim0 = 32
                128,                              // tile_dim1 = 128
                K,                                // tensor_dim0_stride
                0, 0};
      v4i gz = {0, 0, 0, 0};
#if defined(__clang_major__) && __clang_major__ >= 23
      v8i gz8 = {0, 0, 0, 0, 0, 0, 0, 0};
      __builtin_amdgcn_tensor_load_to_lds(g0, g1, gz, gz, gz8, 0);
#else
      __builtin_amdgcn_tensor_load_to_lds(g0, g1, gz, gz, 0);
#endif
#if __has_builtin(__builtin_amdgcn_s_wait_tensorcnt)
      __builtin_amdgcn_s_wait_tensorcnt(0);
#else
      asm volatile("s_wait_tensorcnt 0x0" ::: "memory");
#endif
    }
#else
    // ---- fallback: manual A tile staging ----
    {
      const _Float16* src = A + (size_t)(lb0 + rowS) * K + k0 + colh;
      HalfU u;
      u.q[0] = *(const uint4*)(src);
      u.q[1] = *(const uint4*)(src + 8);
      *(uint4*)&As[rowS * PITCH + colh + 0] = u.q[0];
      *(uint4*)&As[rowS * PITCH + colh + 8] = u.q[1];
    }
#endif
    // ---- stage W tile: 128 x 32, fp32 -> f16 ----
    {
      const float* src = W + (size_t)(ob0 + rowS) * K + k0 + colh;
      const float4 f0 = *(const float4*)(src + 0);
      const float4 f1 = *(const float4*)(src + 4);
      const float4 f2 = *(const float4*)(src + 8);
      const float4 f3 = *(const float4*)(src + 12);
      const float tf[16] = {f0.x, f0.y, f0.z, f0.w, f1.x, f1.y, f1.z, f1.w,
                            f2.x, f2.y, f2.z, f2.w, f3.x, f3.y, f3.z, f3.w};
      HalfU u;
#pragma unroll
      for (int j = 0; j < 16; ++j) u.h[j] = (_Float16)tf[j];
      *(uint4*)&Ws[rowS * PITCH + colh + 0] = u.q[0];
      *(uint4*)&Ws[rowS * PITCH + colh + 8] = u.q[1];
    }
    __syncthreads();

    // ---- load fragments (contiguous 16B ds_load_b128 chunks) ----
    v16h af[2], bf[4];
#pragma unroll
    for (int lt = 0; lt < 2; ++lt) {
      const int lr = wl * 32 + lt * 16 + ln;
      FragU fu;
      fu.q[0] = *(const uint4*)&As[lr * PITCH + 8 * h];        // K in [8h, 8h+8)
      fu.q[1] = *(const uint4*)&As[lr * PITCH + 16 + 8 * h];   // K in [16+8h, ..)
      af[lt] = fu.v;
    }
#pragma unroll
    for (int ot = 0; ot < 4; ++ot) {
      const int orr = wo * 64 + ot * 16 + ln;
      FragU fu;
      fu.q[0] = *(const uint4*)&Ws[orr * PITCH + 8 * h];       // K in [8h, 8h+8)
      fu.q[1] = *(const uint4*)&Ws[orr * PITCH + 16 + 8 * h];  // K in [16+8h, ..)
      bf[ot] = fu.v;
    }

    // ---- 8 WMMAs per wave per K-step ----
#pragma unroll
    for (int lt = 0; lt < 2; ++lt)
#pragma unroll
      for (int ot = 0; ot < 4; ++ot)
        acc[lt][ot] = __builtin_amdgcn_wmma_f32_16x16x32_f16(
            false, af[lt], false, bf[ot], (short)0, acc[lt][ot], false, false);

    __syncthreads();
  }

  // ---- epilogue: store pre-BN Y, accumulate BN stats ----
#pragma unroll
  for (int lt = 0; lt < 2; ++lt) {
#pragma unroll
    for (int ot = 0; ot < 4; ++ot) {
      const int olocal = wo * 64 + ot * 16 + ln;
      const int o = ob0 + olocal;
      float ls = 0.0f, lq = 0.0f;
#pragma unroll
      for (int r = 0; r < 8; ++r) {
        const float vv = acc[lt][ot][r];
        ls += vv;
        lq = fmaf(vv, vv, lq);
        const int l = lb0 + wl * 32 + lt * 16 + r + 8 * h;
        if constexpr (OUT_HALF)
          ((_Float16*)Yout)[(size_t)l * Otot + o] = (_Float16)vv;
        else
          ((float*)Yout)[(size_t)l * Otot + o] = vv;
      }
      atomicAdd(&bsum[olocal], ls);
      atomicAdd(&bsq[olocal], lq);
    }
  }
  __syncthreads();
  if (tid < 128) {
    atomicAdd(&sumO[ob0 + tid], bsum[tid]);
    atomicAdd(&sqO[ob0 + tid], bsq[tid]);
  }
}

// ---------------------------------------------------------------------------
// Fold BN stats into scale/shift:  y_norm = y*scale + shift
// ---------------------------------------------------------------------------
__global__ void bn_params_kernel(const float* __restrict__ sum,
                                 const float* __restrict__ sq,
                                 const float* __restrict__ g,
                                 const float* __restrict__ bb,
                                 float* __restrict__ scale,
                                 float* __restrict__ shift,
                                 const int C, const float invCount) {
  const int c = blockIdx.x * blockDim.x + threadIdx.x;
  if (c < C) {
    const float mean = sum[c] * invCount;
    const float var = fmaf(-mean, mean, sq[c] * invCount);
    const float s = g[c] * rsqrtf(var + 1e-5f);
    scale[c] = s;
    shift[c] = fmaf(-mean, s, bb[c]);
  }
}

// ---------------------------------------------------------------------------
// In-place BN1 + ReLU on Y1 (f16, [L][CMID]); keeps GEMM2's A path a raw copy
// so the TDM can feed it. 16 halfs per thread, channel-aligned chunks.
// ---------------------------------------------------------------------------
__global__ __launch_bounds__(256) void bn_apply_half_kernel(
    _Float16* __restrict__ Y,
    const float* __restrict__ scale,
    const float* __restrict__ shift) {
  const size_t i = ((size_t)blockIdx.x * 256 + threadIdx.x) * 16;
  const int ob = (int)(i & (CMID_ - 1));   // chunk start within 256-ch row
  HalfU u;
  u.q[0] = *(const uint4*)(Y + i);
  u.q[1] = *(const uint4*)(Y + i + 8);
#pragma unroll
  for (int j = 0; j < 16; ++j)
    u.h[j] = (_Float16)fmaxf(
        fmaf((float)u.h[j], scale[ob + j], shift[ob + j]), 0.0f);
  *(uint4*)(Y + i) = u.q[0];
  *(uint4*)(Y + i + 8) = u.q[1];
}

// ---------------------------------------------------------------------------
// Final: Y2[L][COUT] (pre-BN f32) -> out (B, COUT, N) with BN2 + ReLU.
// ---------------------------------------------------------------------------
__global__ __launch_bounds__(256) void out_transpose_kernel(
    const float* __restrict__ Y2,        // [L][COUT]
    const float* __restrict__ scale,
    const float* __restrict__ shift,
    float* __restrict__ out) {           // (B, COUT, N)
  __shared__ float tile[32][33];
  const int lb = blockIdx.x * 32;
  const int ob = blockIdx.y * 32;
  const int tx = threadIdx.x, ty = threadIdx.y;
#pragma unroll
  for (int i = 0; i < 32; i += 8)
    tile[ty + i][tx] = Y2[(size_t)(lb + ty + i) * COUT_ + ob + tx];
  __syncthreads();
  const int b = lb >> 13;
  const int nb = lb & (N_ - 1);
#pragma unroll
  for (int i = 0; i < 32; i += 8) {
    const int o = ob + ty + i;
    const float v = fmaxf(fmaf(tile[tx][ty + i], scale[o], shift[o]), 0.0f);
    out[((size_t)(b * COUT_ + o)) * N_ + nb + tx] = v;
  }
}

// ---------------------------------------------------------------------------
// Host launcher
// ---------------------------------------------------------------------------
extern "C" void kernel_launch(void* const* d_in, const int* in_sizes, int n_in,
                              void* d_out, int out_size, void* d_ws, size_t ws_size,
                              hipStream_t stream) {
  const float* unknown      = (const float*)d_in[0];  // (B, N, 3)
  const float* known        = (const float*)d_in[1];  // (B, M, 3)
  const float* unknow_feats = (const float*)d_in[2];  // (B, C1, N)
  const float* known_feats  = (const float*)d_in[3];  // (B, C2, M)
  const float* W1           = (const float*)d_in[4];  // (CMID, CIN)
  const float* g1           = (const float*)d_in[5];
  const float* b1           = (const float*)d_in[6];
  const float* W2           = (const float*)d_in[7];  // (COUT, CMID)
  const float* g2           = (const float*)d_in[8];
  const float* b2           = (const float*)d_in[9];
  float* out = (float*)d_out;

  // Workspace layout (bytes):
  //   [0)            nn_idx  int   L*3          = 786,432
  //   [786432)       nn_w    float L*3          = 786,432
  //   [1572864)      stats   float 2048 (sums + folded params)
  //   [1581056)      Y1      f16   L*CMID       = 33,554,432
  //   [35135488)     Xcat    f16   L*CIN        = 100,663,296
  //                  Y2      f32   L*COUT (aliases dead Xcat, 64MB <= 100MB)
  char* ws = (char*)d_ws;
  int*   nn_idx = (int*)ws;
  float* nn_w   = (float*)(ws + 786432);
  float* stats  = (float*)(ws + 1572864);
  float* sum1   = stats +    0;
  float* sq1    = stats +  256;
  float* sum2   = stats +  512;
  float* sq2    = stats +  768;
  float* scale1 = stats + 1024;
  float* shift1 = stats + 1280;
  float* scale2 = stats + 1536;
  float* shift2 = stats + 1792;
  _Float16* Y1   = (_Float16*)(ws + 1581056);
  _Float16* Xcat = (_Float16*)(ws + 35135488);
  float*    Y2   = (float*)   (ws + 35135488);

  const float invL = 1.0f / (float)L_;

  // 1) zero BN accumulators (must re-zero every call: atomics accumulate)
  zero_stats_kernel<<<4, 256, 0, stream>>>(stats);

  // 2) 3-NN search
  three_nn_kernel<<<B_ * (N_ / 256), 256, 0, stream>>>(unknown, known, nn_idx, nn_w);

  // 3) build concat features (transposed, f16)
  copy_transpose_kernel<<<dim3(L_ / 32, C1_ / 32), dim3(32, 8), 0, stream>>>(unknow_feats, Xcat);
  interp_concat_kernel<<<L_, 256, 0, stream>>>(known_feats, nn_idx, nn_w, Xcat);

  // 4) GEMM1 (768 -> 256): TDM-fed WMMA, pre-BN f16 output + stats
  gemm_wmma_kernel<true><<<dim3(L_ / 128, CMID_ / 128), 256, GEMM_LDS_BYTES, stream>>>(
      Xcat, W1, CIN_, (void*)Y1, CMID_, sum1, sq1);

  // 5) fold BN1 stats
  bn_params_kernel<<<1, 256, 0, stream>>>(sum1, sq1, g1, b1, scale1, shift1, CMID_, invL);

  // 6) apply BN1 + ReLU in place on Y1 (keeps GEMM2 A-path raw for TDM)
  bn_apply_half_kernel<<<(L_ * CMID_) / (16 * 256), 256, 0, stream>>>(Y1, scale1, shift1);

  // 7) GEMM2 (256 -> 256): TDM-fed WMMA, pre-BN f32 output + stats
  gemm_wmma_kernel<false><<<dim3(L_ / 128, COUT_ / 128), 256, GEMM_LDS_BYTES, stream>>>(
      Y1, W2, CMID_, (void*)Y2, COUT_, sum2, sq2);

  // 8) fold BN2 stats
  bn_params_kernel<<<1, 256, 0, stream>>>(sum2, sq2, g2, b2, scale2, shift2, COUT_, invL);

  // 9) transpose to (B, C, N) with BN2 + ReLU
  out_transpose_kernel<<<dim3(L_ / 32, COUT_ / 32), dim3(32, 8), 0, stream>>>(
      Y2, scale2, shift2, out);
}
